// SparseLinear_71253507441019
// MI455X (gfx1250) — matrix operand order repaired and checked by reading, not compiled, not run on hardware
//
#include <hip/hip_runtime.h>

typedef float          v8f   __attribute__((ext_vector_type(8)));
typedef unsigned int   v8u   __attribute__((ext_vector_type(8)));
typedef __bf16         v16bf __attribute__((ext_vector_type(16)));
typedef unsigned short ushort_t;

#define DIN  4096
#define DOUT 4096

// ---------------- mask preparation kernels ----------------

__global__ __launch_bounds__(256) void k_rowsum(const float* __restrict__ W,
                                                float* __restrict__ rowsum) {
  __shared__ float buf[256];
  const int r = blockIdx.x;
  const float* row = W + (size_t)r * DIN;
  float s = 0.f;
  for (int d = threadIdx.x; d < DIN; d += 256) s += fabsf(row[d]);
  buf[threadIdx.x] = s;
  __syncthreads();
  for (int off = 128; off > 0; off >>= 1) {
    if (threadIdx.x < off) buf[threadIdx.x] += buf[threadIdx.x + off];
    __syncthreads();
  }
  if (threadIdx.x == 0) rowsum[r] = buf[0];
}

// partial column sums over 32 row-segments (deterministic, no atomics)
__global__ __launch_bounds__(256) void k_colpart(const float* __restrict__ W,
                                                 float* __restrict__ part) {
  const int d   = blockIdx.x * 256 + threadIdx.x;
  const int seg = blockIdx.y;
  float s = 0.f;
  const int r0 = seg * (DOUT / 32);
  for (int r = r0; r < r0 + (DOUT / 32); ++r)
    s += fabsf(W[(size_t)r * DIN + d]);
  part[(size_t)seg * DIN + d] = s;
}

__global__ __launch_bounds__(256) void k_colfinal(const float* __restrict__ part,
                                                  float* __restrict__ colsum) {
  const int d = blockIdx.x * 256 + threadIdx.x;
  float s = 0.f;
  for (int seg = 0; seg < 32; ++seg) s += part[(size_t)seg * DIN + d];
  colsum[d] = s;
}

// perm[i][j] = 1 iff j == p[i]; we need pinv[j] = i with p[i] == j
__global__ __launch_bounds__(256) void k_pinv(const float* __restrict__ perm,
                                              int* __restrict__ pinv) {
  const int i = blockIdx.x;
  const float* row = perm + (size_t)i * DIN;
  for (int j = threadIdx.x; j < DIN; j += 256)
    if (row[j] > 0.5f) pinv[j] = i;   // exactly one hit per row
}

__device__ __forceinline__ ushort_t bf16_rne_bits(float f) {
  return __builtin_bit_cast(ushort_t, (__bf16)f);
}

// 2:4 top-k mask in permuted column space; write masked weight split into
// bf16 hi (truncation) / lo (rounded residual) planes for the bf16x3 GEMM.
__global__ __launch_bounds__(256) void k_mask(const float* __restrict__ W,
                                              const float* __restrict__ colsum,
                                              const float* __restrict__ rowsum,
                                              const float* __restrict__ scaler,
                                              const int*   __restrict__ pinv,
                                              ushort_t* __restrict__ Whi,
                                              ushort_t* __restrict__ Wlo) {
  const int r = blockIdx.x;
  const float* wrow = W + (size_t)r * DIN;
  ushort_t* hrow = Whi + (size_t)r * DIN;
  ushort_t* lrow = Wlo + (size_t)r * DIN;
  const float inv_rs = 1.0f / rowsum[r];
  const int nb = DIN / 4;
  for (int b = threadIdx.x; b < nb; b += 256) {
    int dd[4]; float wv[4], mt[4];
    for (int t = 0; t < 4; ++t) {
      const int d = pinv[4 * b + t];
      dd[t] = d;
      const float w = wrow[d];
      wv[t] = w;
      const float aw = fabsf(w);
      // (|w|/colsum + |w|/rowsum) * sqrt(scaler)^0.5 == * scaler^0.25
      mt[t] = (aw / colsum[d] + aw * inv_rs) * sqrtf(sqrtf(scaler[d]));
    }
    // keep top-2 by (value desc, index asc) — matches jax.lax.top_k tie-break
    for (int t = 0; t < 4; ++t) {
      int rank = 0;
      for (int u = 0; u < 4; ++u) {
        if (u == t) continue;
        rank += (mt[u] > mt[t]) || (mt[u] == mt[t] && u < t);
      }
      const float wk = (rank < 2) ? wv[t] : 0.0f;
      const unsigned bits = __float_as_uint(wk);
      const ushort_t hi = (ushort_t)(bits >> 16);               // truncate
      const float tail = wk - __uint_as_float(bits & 0xffff0000u);
      hrow[dd[t]] = hi;
      lrow[dd[t]] = bf16_rne_bits(tail);
    }
  }
}

// split X (f32) into bf16 hi (truncation) / lo (rounded residual) planes
__global__ __launch_bounds__(256) void k_split(const float* __restrict__ X,
                                               ushort_t* __restrict__ Xhi,
                                               ushort_t* __restrict__ Xlo,
                                               int total4) {
  const int stride = gridDim.x * 256;
  for (int i = blockIdx.x * 256 + threadIdx.x; i < total4; i += stride) {
    const float4 v = ((const float4*)X)[i];
    const unsigned b0 = __float_as_uint(v.x), b1 = __float_as_uint(v.y);
    const unsigned b2 = __float_as_uint(v.z), b3 = __float_as_uint(v.w);
    const unsigned h01 = (b0 >> 16) | (b1 & 0xffff0000u);
    const unsigned h23 = (b2 >> 16) | (b3 & 0xffff0000u);
    const unsigned l01 = (unsigned)bf16_rne_bits(v.x - __uint_as_float(b0 & 0xffff0000u))
                       | ((unsigned)bf16_rne_bits(v.y - __uint_as_float(b1 & 0xffff0000u)) << 16);
    const unsigned l23 = (unsigned)bf16_rne_bits(v.z - __uint_as_float(b2 & 0xffff0000u))
                       | ((unsigned)bf16_rne_bits(v.w - __uint_as_float(b3 & 0xffff0000u)) << 16);
    ((uint2*)Xhi)[i] = make_uint2(h01, h23);
    ((uint2*)Xlo)[i] = make_uint2(l01, l23);
  }
}

// ---------------- bf16x3 WMMA GEMM ----------------
// out[m,o] = sum_d x[m,d]*Wm[o,d] + bias[o],  x = xh+xl, w = wh+wl (bf16)
// out ~= xh*wh + xh*wl + xl*wh  (f32 accumulate; xl*wl ~ 2^-16, dropped)

#define BM 128
#define BN 128
#define BK 32
#define ASTR 40   // LDS row stride in bf16 halves (80B: 16B-aligned, conflict-free)

__device__ __forceinline__ v16bf frag_a(const ushort_t* base, int row, int hsel) {
  const ushort_t* p = base + row * ASTR + hsel * 8;   // halves
  uint4 c0 = *(const uint4*)(p);                      // K 0..7 / 8..15
  uint4 c1 = *(const uint4*)(p + 16);                 // K 16..23 / 24..31
  v8u u = {c0.x, c0.y, c0.z, c0.w, c1.x, c1.y, c1.z, c1.w};
  return __builtin_bit_cast(v16bf, u);
}

__device__ __forceinline__ v16bf frag_b(const ushort_t* base, int n, int hsel) {
  const ushort_t* p = base + n * ASTR + hsel * 8;     // K 0..7,16..23 / 8..15,24..31
  uint4 c0 = *(const uint4*)(p);
  uint4 c1 = *(const uint4*)(p + 16);
  v8u u = {c0.x, c0.y, c0.z, c0.w, c1.x, c1.y, c1.z, c1.w};
  return __builtin_bit_cast(v16bf, u);
}

template <bool PRESPLIT>
__global__ __launch_bounds__(256) void k_gemm(const float* __restrict__ X,
                                              const ushort_t* __restrict__ Xhi,
                                              const ushort_t* __restrict__ Xlo,
                                              const ushort_t* __restrict__ Whi,
                                              const ushort_t* __restrict__ Wlo,
                                              const float* __restrict__ bias,
                                              float* __restrict__ Out) {
  __shared__ __align__(16) ushort_t Ah[BM * ASTR];
  __shared__ __align__(16) ushort_t Al[BM * ASTR];
  __shared__ __align__(16) ushort_t Bh[BN * ASTR];
  __shared__ __align__(16) ushort_t Bl[BN * ASTR];

  const int tid  = threadIdx.x;
  const int lane = tid & 31;
  const int wave = tid >> 5;     // 8 waves: 4 in M x 2 in N
  const int wm   = wave & 3;     // M offset wm*32
  const int wn   = wave >> 2;    // N offset wn*64
  const int m0   = blockIdx.y * BM;
  const int n0   = blockIdx.x * BN;
  const int l15  = lane & 15;
  const int hsel = lane >> 4;

  v8f acc[2][4] = {};

  for (int k0 = 0; k0 < DIN; k0 += BK) {
    // ---- stage A: 128 rows x 32 halves (hi/lo) ----
    if constexpr (PRESPLIT) {
      for (int i = 0; i < 2; ++i) {
        const int li = tid + i * 256;        // 512 uint4 per plane
        const int r  = li >> 2;
        const int kq = (li & 3) << 3;
        const size_t g = (size_t)(m0 + r) * DIN + k0 + kq;
        *(uint4*)&Ah[r * ASTR + kq] = *(const uint4*)&Xhi[g];
        *(uint4*)&Al[r * ASTR + kq] = *(const uint4*)&Xlo[g];
      }
    } else {
      for (int i = 0; i < 4; ++i) {
        const int li = tid + i * 256;        // 1024 float4 total
        const int r  = li >> 3;
        const int kq = (li & 7) << 2;
        const float4 v = *(const float4*)&X[(size_t)(m0 + r) * DIN + k0 + kq];
        const unsigned b0 = __float_as_uint(v.x), b1 = __float_as_uint(v.y);
        const unsigned b2 = __float_as_uint(v.z), b3 = __float_as_uint(v.w);
        const unsigned h01 = (b0 >> 16) | (b1 & 0xffff0000u);
        const unsigned h23 = (b2 >> 16) | (b3 & 0xffff0000u);
        const unsigned l01 = (unsigned)bf16_rne_bits(v.x - __uint_as_float(b0 & 0xffff0000u))
                           | ((unsigned)bf16_rne_bits(v.y - __uint_as_float(b1 & 0xffff0000u)) << 16);
        const unsigned l23 = (unsigned)bf16_rne_bits(v.z - __uint_as_float(b2 & 0xffff0000u))
                           | ((unsigned)bf16_rne_bits(v.w - __uint_as_float(b3 & 0xffff0000u)) << 16);
        *(uint2*)&Ah[r * ASTR + kq] = make_uint2(h01, h23);
        *(uint2*)&Al[r * ASTR + kq] = make_uint2(l01, l23);
      }
    }
    // ---- stage B: 128 rows x 32 halves, straight bf16 copies ----
    for (int i = 0; i < 2; ++i) {
      const int li = tid + i * 256;          // 512 uint4 per plane
      const int r  = li >> 2;
      const int kq = (li & 3) << 3;
      const size_t g = (size_t)(n0 + r) * DIN + k0 + kq;
      *(uint4*)&Bh[r * ASTR + kq] = *(const uint4*)&Whi[g];
      *(uint4*)&Bl[r * ASTR + kq] = *(const uint4*)&Wlo[g];
    }
    __syncthreads();

    v16bf ah[2], al[2];
    for (int mi = 0; mi < 2; ++mi) {
      const int row = wm * 32 + mi * 16 + l15;
      ah[mi] = frag_a(Ah, row, hsel);
      al[mi] = frag_a(Al, row, hsel);
    }
    for (int ni = 0; ni < 4; ++ni) {
      const int n = wn * 64 + ni * 16 + l15;
      const v16bf bh = frag_b(Bh, n, hsel);
      const v16bf bl = frag_b(Bl, n, hsel);
      for (int mi = 0; mi < 2; ++mi) {
        acc[mi][ni] = __builtin_amdgcn_wmma_f32_16x16x32_bf16(
            false, ah[mi], false, bh, (short)0, acc[mi][ni], false, false);
        acc[mi][ni] = __builtin_amdgcn_wmma_f32_16x16x32_bf16(
            false, ah[mi], false, bl, (short)0, acc[mi][ni], false, false);
        acc[mi][ni] = __builtin_amdgcn_wmma_f32_16x16x32_bf16(
            false, al[mi], false, bh, (short)0, acc[mi][ni], false, false);
      }
    }
    __syncthreads();
  }

  // C layout: VGPR v holds M = v (lanes 0-15) / v+8 (lanes 16-31), N = lane%16
  for (int mi = 0; mi < 2; ++mi)
    for (int ni = 0; ni < 4; ++ni) {
      const int col = n0 + wn * 64 + ni * 16 + l15;
      const float bv = bias[col];
      const int rowBase = m0 + wm * 32 + mi * 16 + (hsel << 3);
      for (int v = 0; v < 8; ++v)
        Out[(size_t)(rowBase + v) * DOUT + col] = acc[mi][ni][v] + bv;
    }
}

// ---------------- launch ----------------

extern "C" void kernel_launch(void* const* d_in, const int* in_sizes, int n_in,
                              void* d_out, int out_size, void* d_ws, size_t ws_size,
                              hipStream_t stream) {
  const float* x      = (const float*)d_in[0];
  const float* weight = (const float*)d_in[1];
  const float* bias   = (const float*)d_in[2];
  const float* scaler = (const float*)d_in[3];
  const float* perm   = (const float*)d_in[4];
  // d_in[5]=prune_n(2), d_in[6]=prune_m(4): fixed 2:4 pattern

  const int din  = in_sizes[3];            // 4096
  const int dout = in_sizes[2];            // 4096
  const int M    = in_sizes[0] / din;      // 8192

  char* ws = (char*)d_ws;
  ushort_t* Whi = (ushort_t*)ws;  ws += (size_t)dout * din * sizeof(ushort_t);  // 33.5 MB
  ushort_t* Wlo = (ushort_t*)ws;  ws += (size_t)dout * din * sizeof(ushort_t);  // 33.5 MB
  float* colsum = (float*)ws;     ws += (size_t)din  * sizeof(float);
  float* rowsum = (float*)ws;     ws += (size_t)dout * sizeof(float);
  int*   pinv   = (int*)ws;       ws += (size_t)din  * sizeof(int);
  float* part   = (float*)ws;     ws += (size_t)32 * din * sizeof(float);
  ushort_t* Xhi = (ushort_t*)ws;  ws += (size_t)M * din * sizeof(ushort_t);     // 67 MB
  ushort_t* Xlo = (ushort_t*)ws;  ws += (size_t)M * din * sizeof(ushort_t);     // 67 MB
  const bool presplit = ((size_t)(ws - (char*)d_ws) <= ws_size);

  k_rowsum <<<dout, 256, 0, stream>>>(weight, rowsum);
  k_colpart<<<dim3(din / 256, 32), 256, 0, stream>>>(weight, part);
  k_colfinal<<<din / 256, 256, 0, stream>>>(part, colsum);
  k_pinv   <<<din, 256, 0, stream>>>(perm, pinv);
  k_mask   <<<dout, 256, 0, stream>>>(weight, colsum, rowsum, scaler, pinv, Whi, Wlo);

  const dim3 gg(dout / BN, M / BM);
  if (presplit) {
    k_split<<<4096, 256, 0, stream>>>(x, Xhi, Xlo, (M * din) / 4);
    k_gemm<true><<<gg, 256, 0, stream>>>(x, Xhi, Xlo, Whi, Wlo, bias, (float*)d_out);
  } else {
    k_gemm<false><<<gg, 256, 0, stream>>>(x, Xhi, Xlo, Whi, Wlo, bias, (float*)d_out);
  }
}
